// SeperableSelfAttention_33097017983308
// MI455X (gfx1250) — compile-verified, matrix operand order checked by reading, not run on hardware
//
#include <hip/hip_runtime.h>
#include <cstdint>
#include <cstddef>

#define D     384
#define NTOK  4096
#define NGRP  32            // B*P = 8*4
#define TOTTOK (NGRP * NTOK)

typedef __attribute__((ext_vector_type(16))) __bf16 v16bf;
typedef __attribute__((ext_vector_type(8)))  __bf16 v8bf;
typedef __attribute__((ext_vector_type(8)))  float  v8f;

union Frag { v16bf v; v8bf h[2]; };

__device__ __forceinline__ unsigned short f2bf(float f) {
  unsigned int u = __float_as_uint(f);
  unsigned int r = u + 0x7FFFu + ((u >> 16) & 1u);   // round-to-nearest-even
  return (unsigned short)(r >> 16);
}
__device__ __forceinline__ float bf2f(unsigned short h) {
  return __uint_as_float(((unsigned int)h) << 16);
}

// ---------------------------------------------------------------------------
// K0: weights -> bf16, stored K-major (row = output col n, col = k) so that
// B-fragments load with the same two-16B-chunk pattern as A-fragments.
// wTv[n][k] = W_v[k][n] = w_qkv[k*769 + 385 + n]; wTo[n][k] = w_out[k][n]
// ---------------------------------------------------------------------------
__global__ void k0_prep_weights(const float* __restrict__ w_qkv,
                                const float* __restrict__ w_out,
                                unsigned short* __restrict__ wTv,
                                unsigned short* __restrict__ wTo) {
  int idx = blockIdx.x * blockDim.x + threadIdx.x;
  if (idx >= D * D) return;
  int n = idx / D, k = idx % D;
  wTv[n * D + k] = f2bf(w_qkv[(size_t)k * 769 + 385 + n]);
  wTo[n * D + k] = f2bf(w_out[(size_t)k * D + n]);
}

// ---------------------------------------------------------------------------
// K1: q[t] = hidden[t]·w_q + b_q   and   hb = bf16(hidden)
// hidden fp32 is read exactly once in the whole pipeline -> non-temporal
// loads so the 192 MiB stream does not evict hb/weights from the 192 MB L2.
// ---------------------------------------------------------------------------
__global__ void k1_q_and_bf16(const float* __restrict__ hidden,
                              const float* __restrict__ w_qkv,
                              const float* __restrict__ b_qkv,
                              unsigned short* __restrict__ hb,
                              float* __restrict__ q) {
  int lane = threadIdx.x & 31;
  int wid  = threadIdx.x >> 5;
  float wq[12];
#pragma unroll
  for (int i = 0; i < 12; ++i) wq[i] = w_qkv[(size_t)(lane + 32 * i) * 769];
  float b0 = b_qkv[0];
  int gw = blockIdx.x * 8 + wid;
#pragma unroll 1
  for (int s = 0; s < 8; ++s) {
    int t = gw * 8 + s;
    size_t base = (size_t)t * D;
    float acc = 0.f;
#pragma unroll
    for (int i = 0; i < 12; ++i) {
      int d = lane + 32 * i;
      float v = __builtin_nontemporal_load(&hidden[base + d]);
      acc += v * wq[i];
      hb[base + d] = f2bf(v);
    }
#pragma unroll
    for (int off = 16; off > 0; off >>= 1) acc += __shfl_xor(acc, off, 32);
    if (lane == 0) q[t] = acc + b0;
  }
}

// ---------------------------------------------------------------------------
// K2a: softmax stats per group: stats[g] = {max, 1/sum(exp)}
// ---------------------------------------------------------------------------
__global__ void k2a_stats(const float* __restrict__ q, float* __restrict__ stats) {
  __shared__ float red[256];
  int g = blockIdx.x, tid = threadIdx.x;
  const float* qg = q + (size_t)g * NTOK;
  float m = -3.4e38f;
  for (int i = tid; i < NTOK; i += 256) m = fmaxf(m, qg[i]);
  red[tid] = m; __syncthreads();
  for (int o = 128; o > 0; o >>= 1) {
    if (tid < o) red[tid] = fmaxf(red[tid], red[tid + o]);
    __syncthreads();
  }
  m = red[0]; __syncthreads();
  float s = 0.f;
  for (int i = tid; i < NTOK; i += 256) s += expf(qg[i] - m);
  red[tid] = s; __syncthreads();
  for (int o = 128; o > 0; o >>= 1) {
    if (tid < o) red[tid] += red[tid + o];
    __syncthreads();
  }
  if (tid == 0) { stats[g * 2] = m; stats[g * 2 + 1] = 1.f / red[0]; }
}

// ---------------------------------------------------------------------------
// K2b: deterministic partial sums of hbar = sum_i softmax_i * h_i
// grid = NGRP*32 blocks x 384 threads; each block: 128-token chunk.
// Reads hb (regular temporal -> primes/keeps it in L2 for K4).
// ---------------------------------------------------------------------------
__global__ void k2b_hbar_partial(const unsigned short* __restrict__ hb,
                                 const float* __restrict__ q,
                                 const float* __restrict__ stats,
                                 float* __restrict__ partial) {
  __shared__ float sp[128];
  int g = blockIdx.x >> 5;
  int c = blockIdx.x & 31;
  int tid = threadIdx.x;
  int t0 = c * 128;
  float m = stats[g * 2], inv = stats[g * 2 + 1];
  if (tid < 128) sp[tid] = expf(q[(size_t)g * NTOK + t0 + tid] - m) * inv;
  __syncthreads();
  int d = tid;                               // 384 threads, one per dim
  const unsigned short* hg = hb + ((size_t)g * NTOK + t0) * D;
  float acc = 0.f;
  for (int i = 0; i < 128; ++i) acc += sp[i] * bf2f(hg[(size_t)i * D + d]);
  partial[((size_t)g * 32 + c) * D + d] = acc;
}

__global__ void k2c_hbar_reduce(const float* __restrict__ partial,
                                float* __restrict__ hbar) {
  int g = blockIdx.x, d = threadIdx.x;
  float acc = 0.f;
  for (int c = 0; c < 32; ++c) acc += partial[((size_t)g * 32 + c) * D + d];
  hbar[g * D + d] = acc;
}

// ---------------------------------------------------------------------------
// K3: cv[g] = hbar[g] @ W_k + b_k   (linearity: cv = (sum s_i h_i) W_k + b_k)
// ---------------------------------------------------------------------------
__global__ void k3_cv(const float* __restrict__ hbar,
                      const float* __restrict__ w_qkv,
                      const float* __restrict__ b_qkv,
                      float* __restrict__ cv) {
  int g = blockIdx.x, n = threadIdx.x;
  const float* hg = hbar + g * D;
  float acc = b_qkv[1 + n];
  for (int dd = 0; dd < D; ++dd) acc += hg[dd] * w_qkv[(size_t)dd * 769 + 1 + n];
  cv[g * D + n] = acc;
}

// ---------------------------------------------------------------------------
// K4: main WMMA kernel.
//   V  = H_tile @ W_v + b_v            (bf16 WMMA, f32 accum)
//   E  = cv * relu(V)       -> bf16 -> same LDS tile
//   O  = E @ W_out + b_out   -> fp32 non-temporal store (write-once stream;
//                               keeps hb + weights resident in L2)
// 256 threads = 8 waves; tile = 64 tokens x 384; wave owns a 16-row block and
// 12 of the 24 16-col blocks. 12 k-steps of v_wmma_f32_16x16x32_bf16 each.
// ---------------------------------------------------------------------------
__global__ void __launch_bounds__(256)
k4_main(const unsigned short* __restrict__ hb,
        const unsigned short* __restrict__ wTv,
        const unsigned short* __restrict__ wTo,
        const float* __restrict__ b_qkv,
        const float* __restrict__ b_out,
        const float* __restrict__ cv,
        float* __restrict__ out) {
  __shared__ unsigned short sh[64 * D];          // 48 KB tile
  int g   = blockIdx.x >> 6;
  int tb  = blockIdx.x & 63;
  int row0 = tb * 64;
  size_t tokbase = (size_t)g * NTOK + row0;
  int tid = threadIdx.x;

  // stage bf16 H tile into LDS (16B vector copies; should hit L2 after K2b)
  {
    const uint4* src = (const uint4*)(hb + tokbase * D);
    uint4* dst = (uint4*)sh;
    for (int i = tid; i < (64 * D) / 8; i += 256) dst[i] = src[i];
  }
  __syncthreads();

  int lane = tid & 31;
  int wid  = tid >> 5;
  int ln   = lane & 15;
  int half = lane >> 4;
  int rb   = wid >> 1;            // 16-row block 0..3
  int cb0  = (wid & 1) * 12;      // first of 12 col-blocks
  int arow = rb * 16 + ln;

  const v8f vzero = {0, 0, 0, 0, 0, 0, 0, 0};
  v8f acc[12];
#pragma unroll
  for (int j = 0; j < 12; ++j) acc[j] = vzero;

  // ---- GEMM1: V = H @ W_v ----
#pragma unroll 1
  for (int ks = 0; ks < 12; ++ks) {
    int kb = ks * 32;
    Frag A;
    A.h[0] = *(const v8bf*)(&sh[arow * D + kb + half * 8]);
    A.h[1] = *(const v8bf*)(&sh[arow * D + kb + 16 + half * 8]);
#pragma unroll
    for (int j = 0; j < 12; ++j) {
      int brow = (cb0 + j) * 16 + ln;
      Frag B;
      B.h[0] = *(const v8bf*)(wTv + (size_t)brow * D + kb + half * 8);
      B.h[1] = *(const v8bf*)(wTv + (size_t)brow * D + kb + 16 + half * 8);
      acc[j] = __builtin_amdgcn_wmma_f32_16x16x32_bf16(
          false, A.v, false, B.v, (short)0, acc[j], false, false);
    }
  }
  __syncthreads();   // all waves done reading H from LDS

  // ---- epilogue1: E = cv * relu(V + b_v) -> bf16 into same LDS tile ----
#pragma unroll
  for (int j = 0; j < 12; ++j) {
    int n = (cb0 + j) * 16 + ln;
    float bv  = b_qkv[385 + n];
    float cvn = cv[g * D + n];
#pragma unroll
    for (int r = 0; r < 8; ++r) {
      int mrow = rb * 16 + half * 8 + r;        // D-matrix VGPR layout
      float v = acc[j][r] + bv;
      v = fmaxf(v, 0.f) * cvn;
      sh[mrow * D + n] = f2bf(v);
    }
  }
  __syncthreads();

  // ---- GEMM2: O = E @ W_out ----
#pragma unroll
  for (int j = 0; j < 12; ++j) acc[j] = vzero;
#pragma unroll 1
  for (int ks = 0; ks < 12; ++ks) {
    int kb = ks * 32;
    Frag A;
    A.h[0] = *(const v8bf*)(&sh[arow * D + kb + half * 8]);
    A.h[1] = *(const v8bf*)(&sh[arow * D + kb + 16 + half * 8]);
#pragma unroll
    for (int j = 0; j < 12; ++j) {
      int brow = (cb0 + j) * 16 + ln;
      Frag B;
      B.h[0] = *(const v8bf*)(wTo + (size_t)brow * D + kb + half * 8);
      B.h[1] = *(const v8bf*)(wTo + (size_t)brow * D + kb + 16 + half * 8);
      acc[j] = __builtin_amdgcn_wmma_f32_16x16x32_bf16(
          false, A.v, false, B.v, (short)0, acc[j], false, false);
    }
  }

  // ---- epilogue2: out = O + b_out (non-temporal stores) ----
#pragma unroll
  for (int j = 0; j < 12; ++j) {
    int n = (cb0 + j) * 16 + ln;
    float bo = b_out[n];
#pragma unroll
    for (int r = 0; r < 8; ++r) {
      int mrow = rb * 16 + half * 8 + r;
      __builtin_nontemporal_store(acc[j][r] + bo, &out[(tokbase + mrow) * D + n]);
    }
  }
}

// ---------------------------------------------------------------------------
extern "C" void kernel_launch(void* const* d_in, const int* in_sizes, int n_in,
                              void* d_out, int out_size, void* d_ws, size_t ws_size,
                              hipStream_t stream) {
  const float* hidden = (const float*)d_in[0];
  const float* w_qkv  = (const float*)d_in[1];
  const float* b_qkv  = (const float*)d_in[2];
  const float* w_out  = (const float*)d_in[3];
  const float* b_out  = (const float*)d_in[4];
  float* out = (float*)d_out;
  (void)in_sizes; (void)n_in; (void)out_size; (void)ws_size;

  char* ws = (char*)d_ws;
  size_t off = 0;
  auto alloc = [&](size_t bytes) -> void* {
    void* p = ws + off;
    off = (off + bytes + 255) & ~(size_t)255;
    return p;
  };
  unsigned short* hb   = (unsigned short*)alloc((size_t)TOTTOK * D * 2); // 96 MiB
  float* q             = (float*)alloc((size_t)TOTTOK * 4);
  float* stats         = (float*)alloc((size_t)NGRP * 2 * 4);
  float* partial       = (float*)alloc((size_t)NGRP * 32 * D * 4);
  float* hbar          = (float*)alloc((size_t)NGRP * D * 4);
  float* cvb           = (float*)alloc((size_t)NGRP * D * 4);
  unsigned short* wTv  = (unsigned short*)alloc((size_t)D * D * 2);
  unsigned short* wTo  = (unsigned short*)alloc((size_t)D * D * 2);

  k0_prep_weights<<<(D * D + 255) / 256, 256, 0, stream>>>(w_qkv, w_out, wTv, wTo);
  k1_q_and_bf16<<<TOTTOK / 64, 256, 0, stream>>>(hidden, w_qkv, b_qkv, hb, q);
  k2a_stats<<<NGRP, 256, 0, stream>>>(q, stats);
  k2b_hbar_partial<<<NGRP * 32, 384, 0, stream>>>(hb, q, stats, partial);
  k2c_hbar_reduce<<<NGRP, 384, 0, stream>>>(partial, hbar);
  k3_cv<<<NGRP, 384, 0, stream>>>(hbar, w_qkv, b_qkv, cvb);
  k4_main<<<NGRP * 64, 256, 0, stream>>>(hb, wTv, wTo, b_qkv, b_out, cvb, out);
}
